// CalibCLIP_8924942041467
// MI455X (gfx1250) — matrix operand order repaired and checked by reading, not compiled, not run on hardware
//
#include <hip/hip_runtime.h>

// ---------------------------------------------------------------------------
// CLIP ViT encoder for MI455X (gfx1250).
// GEMMs via v_wmma_f32_16x16x32_bf16; A-tile staging via the Tensor Data
// Mover (tensor_load_to_lds + s_wait_tensorcnt) when the builtin exists.
// ---------------------------------------------------------------------------

typedef __attribute__((ext_vector_type(16))) __bf16 v16bf;
typedef __attribute__((ext_vector_type(8)))  float  v8f;
typedef __attribute__((ext_vector_type(4)))  unsigned int v4u;
typedef __attribute__((ext_vector_type(4)))  int    v4i;
typedef __attribute__((ext_vector_type(8)))  int    v8i;

#define BATCH 64
#define DMODEL 768
#define NLAYER 12
#define NHEAD 12
#define DHEAD 64
#define DFFN 3072
#define SEQN 193
#define NPATCH 192
#define OUTD 512

#if defined(__gfx1250__) && __has_builtin(__builtin_amdgcn_tensor_load_to_lds)
#define HAVE_TDM 1
#else
#define HAVE_TDM 0
#endif

// ---- bf16 <-> f32 helpers (bit-level) -------------------------------------
static __device__ __forceinline__ __bf16 f2bf(float f) {
    union { float f; unsigned u; } x; x.f = f;
    unsigned r = x.u + 0x7fffu + ((x.u >> 16) & 1u);   // round-to-nearest-even
    unsigned short h = (unsigned short)(r >> 16);
    return __builtin_bit_cast(__bf16, h);
}
static __device__ __forceinline__ float bf2f(__bf16 b) {
    unsigned short h = __builtin_bit_cast(unsigned short, b);
    union { unsigned u; float f; } x; x.u = ((unsigned)h) << 16;
    return x.f;
}

// ---------------------------------------------------------------------------
// Generic bf16 GEMM: C[M,N] = act(A[M,K] @ B[K,N] + bias) (+ residual)
// 256 threads = 8 wave32s. Workgroup tile 128x128, wave tile 32x64
// (2x4 WMMA), K-step 32. LDS rows padded to 40 bf16 (80B) for bank spread.
// Requires: K % 32 == 0, N % 128 == 0 (true for every call site).
// ---------------------------------------------------------------------------
#define BM 128
#define BN 128
#define BKK 32
#define LDP 40   // padded LDS row length (bf16 elements), 80 bytes

template<bool BIAS, bool GELU, bool RES, bool BF16OUT>
__global__ __launch_bounds__(256)
void gemm_bf16_kernel(const __bf16* __restrict__ A, const __bf16* __restrict__ Bw,
                      const float* __restrict__ bias, const float* __restrict__ res,
                      float* __restrict__ Cf, __bf16* __restrict__ Cb,
                      int M, int N, int K)
{
    __shared__ __attribute__((aligned(16))) __bf16 As[BM][LDP];  // 10 KB
    __shared__ __attribute__((aligned(16))) __bf16 Bs[BN][LDP];  // 10 KB ([n][k])

    const int tid  = threadIdx.x;
    const int lane = tid & 31;
    const int wave = tid >> 5;            // 0..7
    const int wm   = (wave & 3) * 32;     // wave M offset inside tile
    const int wn   = (wave >> 2) * 64;    // wave N offset inside tile
    const int m0   = blockIdx.y * BM;
    const int n0   = blockIdx.x * BN;
    const int g    = lane >> 4;           // lane half (WMMA layout)
    const int lr   = lane & 15;

    v8f acc[2][4];
    #pragma unroll
    for (int i = 0; i < 2; ++i)
        #pragma unroll
        for (int j = 0; j < 4; ++j)
            #pragma unroll
            for (int e = 0; e < 8; ++e) acc[i][j][e] = 0.0f;

    for (int k0 = 0; k0 < K; k0 += BKK) {
#if HAVE_TDM
        // ---- A tile via Tensor Data Mover: [BM x BKK] bf16 sub-tile of the
        // row-major [M,K] tensor -> LDS rows padded to 80B (pad 4 dwords
        // every 16 dwords). OOB rows (M edge) return zero per TDM semantics.
        if (wave == 0) {
            unsigned lds_off = (unsigned)(size_t)(&As[0][0]);
            unsigned long long ga =
                (unsigned long long)(const void*)(A + (size_t)m0 * K + k0);
            v4u g0;
            g0[0] = 1u;                                   // count=1, user D#
            g0[1] = lds_off;                              // lds_addr
            g0[2] = (unsigned)(ga & 0xFFFFFFFFu);         // global_addr lo
            g0[3] = (unsigned)((ga >> 32) & 0x01FFFFFFu)  // global_addr hi
                    | (2u << 30);                         // type = 2 (image)
            unsigned dim0 = (unsigned)(K - k0);           // remaining cols
            unsigned dim1 = (unsigned)(M - m0);           // remaining rows
            v8i g1;
            g1[0] = (int)(0x00010000u                     // data_size=1 (2B)
                          | (1u << 20)                    // pad_enable
                          | (3u << 22)                    // pad_interval: 16 dw
                          | (3u << 25));                  // pad_amount: 4 dw
            g1[1] = (int)((dim0 & 0xFFFFu) << 16);        // tensor_dim0 lo
            g1[2] = (int)((dim0 >> 16) | ((dim1 & 0xFFFFu) << 16));
            g1[3] = (int)((dim1 >> 16) | ((unsigned)BKK << 16)); // tile_dim0
            g1[4] = (int)BM;                              // tile_dim1 (dim2=0)
            g1[5] = (int)(unsigned)K;                     // dim0_stride lo
            g1[6] = 0;
            g1[7] = 0;
            v4i z4 = {0, 0, 0, 0};
#if __clang_major__ >= 23
            v8i z8 = {0, 0, 0, 0, 0, 0, 0, 0};
            __builtin_amdgcn_tensor_load_to_lds(g0, g1, z4, z4, z8, 0);
#else
            __builtin_amdgcn_tensor_load_to_lds(g0, g1, z4, z4, 0);
#endif
        }
#else
        // ---- A tile (128x32): 8 bf16 (16B) per thread, two passes.
        // Clamp row instead of branching: out-of-range A rows only feed
        // out-of-range C rows, which the epilogue masks.
        {
            int r = tid >> 2;
            int c = (tid & 3) * 8;
            #pragma unroll
            for (int half = 0; half < 2; ++half) {
                int row  = r + half * 64;
                int grow = m0 + row;
                grow = (grow < M) ? grow : (M - 1);
                const __bf16* src = A + (size_t)grow * K + k0 + c;
                *(uint4*)&As[row][c] = *(const uint4*)src;
            }
        }
#endif
        // ---- B tile (32x128) transposed into Bs[n][k] ----
        {
            int k  = tid >> 3;           // 0..31
            int nv = (tid & 7) * 8;      // 0..56
            #pragma unroll
            for (int half = 0; half < 2; ++half) {
                int n = nv + half * 64;
                const __bf16* src = Bw + (size_t)(k0 + k) * N + n0 + n;
                __bf16 tmp[8];
                *(uint4*)tmp = *(const uint4*)src;
                #pragma unroll
                for (int i = 0; i < 8; ++i) Bs[n + i][k] = tmp[i];
            }
        }
        // ---- prefetch next K-tile while this one is consumed ----
        if (k0 + BKK < K) {
            int pr = m0 + (tid >> 1);
            pr = (pr < M) ? pr : (M - 1);
            __builtin_prefetch(A + (size_t)pr * K + k0 + BKK + (tid & 1) * 16, 0, 3);
            __builtin_prefetch(Bw + (size_t)(k0 + BKK + (tid >> 3)) * N
                                  + n0 + (tid & 7) * 16, 0, 3);
        }
#if HAVE_TDM
        if (wave == 0) __builtin_amdgcn_s_wait_tensorcnt(0);
#endif
        __syncthreads();

        // ---- WMMA fragments per ISA 16-bit layouts ----
        v16bf afrag[2], bfrag[4];
        #pragma unroll
        for (int i = 0; i < 2; ++i) {
            int row = wm + i * 16 + lr;
            #pragma unroll
            for (int v = 0; v < 8; ++v) {
                int kk = ((v < 4) ? 0 : 16) + g * 8 + (v & 3) * 2;
                afrag[i][2 * v]     = As[row][kk];
                afrag[i][2 * v + 1] = As[row][kk + 1];
            }
        }
        #pragma unroll
        for (int j = 0; j < 4; ++j) {
            int col = wn + j * 16 + lr;
            #pragma unroll
            for (int v = 0; v < 8; ++v) {
                int kk = ((v < 4) ? 0 : 16) + g * 8 + (v & 3) * 2;
                bfrag[j][2 * v]     = Bs[col][kk];
                bfrag[j][2 * v + 1] = Bs[col][kk + 1];
            }
        }
        #pragma unroll
        for (int i = 0; i < 2; ++i)
            #pragma unroll
            for (int j = 0; j < 4; ++j)
                acc[i][j] = __builtin_amdgcn_wmma_f32_16x16x32_bf16(
                    false, afrag[i], false, bfrag[j],
                    (short)0, acc[i][j], false, false);
        __syncthreads();
    }

    // ---- epilogue: C layout = 8 VGPRs, row = base + g*8 + r, col = base + lr
    #pragma unroll
    for (int i = 0; i < 2; ++i) {
        #pragma unroll
        for (int j = 0; j < 4; ++j) {
            int col = n0 + wn + j * 16 + lr;
            #pragma unroll
            for (int r = 0; r < 8; ++r) {
                int row = m0 + wm + i * 16 + g * 8 + r;
                if (row < M) {
                    float v = acc[i][j][r];
                    if (BIAS) v += bias[col];
                    if (GELU) v = v * (1.0f / (1.0f + __expf(-1.702f * v)));
                    if (RES)  v += res[(size_t)row * N + col];
                    if (BF16OUT) Cb[(size_t)row * N + col] = f2bf(v);
                    else         Cf[(size_t)row * N + col] = v;
                }
            }
        }
    }
}

// ---------------------------------------------------------------------------
// f32 -> bf16 bulk conversion (weights), grid-stride
// ---------------------------------------------------------------------------
__global__ __launch_bounds__(256)
void f2bf_kernel(const float* __restrict__ in, __bf16* __restrict__ out, size_t n)
{
    size_t i = (size_t)blockIdx.x * 256 + threadIdx.x;
    size_t stride = (size_t)gridDim.x * 256;
    for (; i < n; i += stride) out[i] = f2bf(in[i]);
}

// bf16 -> f32 widen
__global__ __launch_bounds__(256)
void widen_kernel(const __bf16* __restrict__ in, float* __restrict__ out, size_t n)
{
    size_t i = (size_t)blockIdx.x * 256 + threadIdx.x;
    size_t stride = (size_t)gridDim.x * 256;
    for (; i < n; i += stride) out[i] = bf2f(in[i]);
}

// ---------------------------------------------------------------------------
// im2col for stride==patch conv: Xp[b*192 + ph*8 + pw][(h*16+w)*3 + c]
// ---------------------------------------------------------------------------
__global__ __launch_bounds__(256)
void im2col_kernel(const float* __restrict__ img, __bf16* __restrict__ Xp)
{
    int row = blockIdx.x;                 // 0 .. B*NPATCH-1
    int b = row / NPATCH, p = row % NPATCH;
    int ph = p / 8, pw = p % 8;
    for (int e = threadIdx.x; e < DMODEL; e += 256) {
        int c = e % 3, w = (e / 3) % 16, h = e / 48;
        float v = img[(((size_t)b * 3 + c) * 384 + (size_t)(ph * 16 + h)) * 128
                      + pw * 16 + w];
        Xp[(size_t)row * DMODEL + e] = f2bf(v);
    }
}

// ---------------------------------------------------------------------------
// x = concat(cls, patches) + pos   (f32 residual stream)
// ---------------------------------------------------------------------------
__global__ __launch_bounds__(256)
void assemble_kernel(const float* __restrict__ Pp, const float* __restrict__ cls,
                     const float* __restrict__ pos, float* __restrict__ x)
{
    int row = blockIdx.x;                 // 0 .. B*SEQ-1
    int b = row / SEQN, s = row % SEQN;
    for (int e = threadIdx.x; e < DMODEL; e += 256) {
        float v = (s == 0) ? cls[e]
                           : Pp[((size_t)b * NPATCH + (s - 1)) * DMODEL + e];
        x[(size_t)row * DMODEL + e] = v + pos[(size_t)s * DMODEL + e];
    }
}

// ---------------------------------------------------------------------------
// LayerNorm over D=768, output bf16 (GEMM A input)
// ---------------------------------------------------------------------------
__global__ __launch_bounds__(256)
void layernorm_kernel(const float* __restrict__ x, const float* __restrict__ gw,
                      const float* __restrict__ bw, __bf16* __restrict__ out)
{
    int row = blockIdx.x;
    const float* xr = x + (size_t)row * DMODEL;
    __shared__ float red[256];

    float s = 0.0f;
    for (int i = threadIdx.x; i < DMODEL; i += 256) s += xr[i];
    red[threadIdx.x] = s; __syncthreads();
    for (int o = 128; o > 0; o >>= 1) {
        if (threadIdx.x < o) red[threadIdx.x] += red[threadIdx.x + o];
        __syncthreads();
    }
    float mean = red[0] * (1.0f / DMODEL);
    __syncthreads();

    float vs = 0.0f;
    for (int i = threadIdx.x; i < DMODEL; i += 256) {
        float d = xr[i] - mean; vs += d * d;
    }
    red[threadIdx.x] = vs; __syncthreads();
    for (int o = 128; o > 0; o >>= 1) {
        if (threadIdx.x < o) red[threadIdx.x] += red[threadIdx.x + o];
        __syncthreads();
    }
    float rstd = rsqrtf(red[0] * (1.0f / DMODEL) + 1e-5f);

    for (int i = threadIdx.x; i < DMODEL; i += 256)
        out[(size_t)row * DMODEL + i] = f2bf((xr[i] - mean) * rstd * gw[i] + bw[i]);
}

// ---------------------------------------------------------------------------
// Attention per (b, head): two-pass softmax (max, then exp-sum), K/V staged in LDS (bf16, ~49 KB).
// ~4% of total FLOPs -> VALU f32 is the right cost/complexity point.
// For the last layer, the thread handling q-row 0 stores normalized probs.
// ---------------------------------------------------------------------------
__global__ __launch_bounds__(256)
void attention_kernel(const __bf16* __restrict__ qkv, __bf16* __restrict__ out,
                      float* __restrict__ probs0, int last)
{
    int b = blockIdx.x / NHEAD, h = blockIdx.x % NHEAD;
    __shared__ __bf16 Ks[SEQN][DHEAD];
    __shared__ __bf16 Vs[SEQN][DHEAD];

    for (int i = threadIdx.x; i < SEQN * DHEAD; i += 256) {
        int s = i / DHEAD, d = i % DHEAD;
        size_t base = ((size_t)(b * SEQN + s)) * (3 * DMODEL) + h * DHEAD + d;
        Ks[s][d] = qkv[base + DMODEL];
        Vs[s][d] = qkv[base + 2 * DMODEL];
    }
    __syncthreads();

    const float scale = 0.125f;  // 1/sqrt(64)
    for (int s = threadIdx.x; s < SEQN; s += 256) {
        float q[DHEAD];
        size_t qbase = ((size_t)(b * SEQN + s)) * (3 * DMODEL) + h * DHEAD;
        #pragma unroll
        for (int d = 0; d < DHEAD; ++d) q[d] = bf2f(qkv[qbase + d]) * scale;

        float mx = -1e30f;
        for (int k = 0; k < SEQN; ++k) {
            float sc = 0.0f;
            #pragma unroll
            for (int d = 0; d < DHEAD; ++d) sc += q[d] * bf2f(Ks[k][d]);
            mx = fmaxf(mx, sc);
        }
        float sum = 0.0f;
        float o[DHEAD];
        #pragma unroll
        for (int d = 0; d < DHEAD; ++d) o[d] = 0.0f;
        for (int k = 0; k < SEQN; ++k) {
            float sc = 0.0f;
            #pragma unroll
            for (int d = 0; d < DHEAD; ++d) sc += q[d] * bf2f(Ks[k][d]);
            float p = __expf(sc - mx);
            sum += p;
            #pragma unroll
            for (int d = 0; d < DHEAD; ++d) o[d] += p * bf2f(Vs[k][d]);
        }
        float inv = 1.0f / sum;
        size_t obase = ((size_t)(b * SEQN + s)) * DMODEL + h * DHEAD;
        #pragma unroll
        for (int d = 0; d < DHEAD; ++d) out[obase + d] = f2bf(o[d] * inv);

        if (last && s == 0) {
            for (int k = 1; k < SEQN; ++k) {
                float sc = 0.0f;
                #pragma unroll
                for (int d = 0; d < DHEAD; ++d) sc += q[d] * bf2f(Ks[k][d]);
                probs0[((size_t)b * NHEAD + h) * NPATCH + (k - 1)] =
                    __expf(sc - mx) * inv;
            }
        }
    }
}

// ---------------------------------------------------------------------------
__global__ __launch_bounds__(256)
void gather_cls_kernel(const __bf16* __restrict__ xn, __bf16* __restrict__ out)
{
    int b = blockIdx.x;
    for (int e = threadIdx.x; e < DMODEL; e += 256)
        out[(size_t)b * DMODEL + e] = xn[((size_t)b * SEQN) * DMODEL + e];
}

__global__ __launch_bounds__(256)
void normalize_kernel(float* __restrict__ feats)
{
    int b = blockIdx.x;
    __shared__ float red[256];
    float s = 0.0f;
    for (int i = threadIdx.x; i < OUTD; i += 256) {
        float v = feats[(size_t)b * OUTD + i]; s += v * v;
    }
    red[threadIdx.x] = s; __syncthreads();
    for (int o = 128; o > 0; o >>= 1) {
        if (threadIdx.x < o) red[threadIdx.x] += red[threadIdx.x + o];
        __syncthreads();
    }
    float inv = rsqrtf(red[0]);
    for (int i = threadIdx.x; i < OUTD; i += 256)
        feats[(size_t)b * OUTD + i] *= inv;
}

__global__ __launch_bounds__(256)
void head_mean_kernel(const float* __restrict__ probs, float* __restrict__ outp)
{
    int i = blockIdx.x * 256 + threadIdx.x;       // over B*NPATCH
    if (i >= BATCH * NPATCH) return;
    int b = i / NPATCH, k = i % NPATCH;
    float s = 0.0f;
    for (int h = 0; h < NHEAD; ++h)
        s += probs[((size_t)b * NHEAD + h) * NPATCH + k];
    outp[i] = s * (1.0f / NHEAD);
}

// ---------------------------------------------------------------------------
static inline int ceildiv(int a, int b) { return (a + b - 1) / b; }

extern "C" void kernel_launch(void* const* d_in, const int* in_sizes, int n_in,
                              void* d_out, int out_size, void* d_ws, size_t ws_size,
                              hipStream_t stream)
{
    (void)in_sizes; (void)n_in; (void)out_size; (void)ws_size;

    const float* images    = (const float*)d_in[0];
    const float* conv_w    = (const float*)d_in[1];
    const float* cls       = (const float*)d_in[2];
    const float* pos       = (const float*)d_in[3];
    const float* ln_pre_g  = (const float*)d_in[4];
    const float* ln_pre_b  = (const float*)d_in[5];
    const float* ln1_g     = (const float*)d_in[6];
    const float* ln1_b     = (const float*)d_in[7];
    const float* w_qkv     = (const float*)d_in[8];
    const float* b_qkv     = (const float*)d_in[9];
    const float* w_o       = (const float*)d_in[10];
    const float* b_o       = (const float*)d_in[11];
    const float* ln2_g     = (const float*)d_in[12];
    const float* ln2_b     = (const float*)d_in[13];
    const float* w1        = (const float*)d_in[14];
    const float* b1        = (const float*)d_in[15];
    const float* w2        = (const float*)d_in[16];
    const float* b2        = (const float*)d_in[17];
    const float* ln_post_g = (const float*)d_in[18];
    const float* ln_post_b = (const float*)d_in[19];
    const float* proj      = (const float*)d_in[20];

    const int ROWS = BATCH * SEQN;     // 12352
    const int PROWS = BATCH * NPATCH;  // 12288

    // ---- workspace carve ----
    char* ws = (char*)d_ws;
    size_t off = 0;
    auto carve = [&](size_t bytes) -> void* {
        off = (off + 255) & ~(size_t)255;
        void* p = ws + off;
        off += bytes;
        return p;
    };
    __bf16* wconv_bf = (__bf16*)carve((size_t)DMODEL * DMODEL * 2);
    __bf16* wqkv_bf  = (__bf16*)carve((size_t)NLAYER * DMODEL * 3 * DMODEL * 2);
    __bf16* wo_bf    = (__bf16*)carve((size_t)NLAYER * DMODEL * DMODEL * 2);
    __bf16* w1_bf    = (__bf16*)carve((size_t)NLAYER * DMODEL * DFFN * 2);
    __bf16* w2_bf    = (__bf16*)carve((size_t)NLAYER * DFFN * DMODEL * 2);
    __bf16* wproj_bf = (__bf16*)carve((size_t)DMODEL * OUTD * 2);
    float*  x        = (float*) carve((size_t)ROWS * DMODEL * 4);
    __bf16* xn       = (__bf16*)carve((size_t)ROWS * DMODEL * 2);  // also im2col Xp
    __bf16* qkvb     = (__bf16*)carve((size_t)ROWS * 3 * DMODEL * 2);
    __bf16* attnb    = (__bf16*)carve((size_t)ROWS * DMODEL * 2);
    __bf16* hb       = (__bf16*)carve((size_t)ROWS * DFFN * 2);
    float*  probs    = (float*) carve((size_t)BATCH * NHEAD * NPATCH * 4);
    __bf16* acls     = (__bf16*)carve((size_t)BATCH * DMODEL * 2);
    float*  Pp       = (float*)hb;  // patch-embed f32 scratch, overlapped with hb

    float* feats    = (float*)d_out;                    // [64,512]
    float* cls_attn = (float*)d_out + BATCH * OUTD;     // [64,192]

    // ---- convert weights to bf16 ----
    f2bf_kernel<<<2048, 256, 0, stream>>>(conv_w, wconv_bf, (size_t)DMODEL * DMODEL);
    f2bf_kernel<<<4096, 256, 0, stream>>>(w_qkv, wqkv_bf, (size_t)NLAYER * DMODEL * 3 * DMODEL);
    f2bf_kernel<<<4096, 256, 0, stream>>>(w_o,   wo_bf,   (size_t)NLAYER * DMODEL * DMODEL);
    f2bf_kernel<<<4096, 256, 0, stream>>>(w1,    w1_bf,   (size_t)NLAYER * DMODEL * DFFN);
    f2bf_kernel<<<4096, 256, 0, stream>>>(w2,    w2_bf,   (size_t)NLAYER * DFFN * DMODEL);
    f2bf_kernel<<<1024, 256, 0, stream>>>(proj,  wproj_bf, (size_t)DMODEL * OUTD);

    // ---- patch embedding: im2col + GEMM [12288,768] x [768,768] ----
    im2col_kernel<<<PROWS, 256, 0, stream>>>(images, xn);
    gemm_bf16_kernel<false, false, false, false>
        <<<dim3(DMODEL / BN, ceildiv(PROWS, BM)), 256, 0, stream>>>(
            xn, wconv_bf, nullptr, nullptr, Pp, nullptr, PROWS, DMODEL, DMODEL);

    // ---- x = [cls; patches] + pos ; x <- ln_pre(x) ----
    assemble_kernel<<<ROWS, 256, 0, stream>>>(Pp, cls, pos, x);
    layernorm_kernel<<<ROWS, 256, 0, stream>>>(x, ln_pre_g, ln_pre_b, xn);
    widen_kernel<<<4096, 256, 0, stream>>>(xn, x, (size_t)ROWS * DMODEL);

    // ---- transformer layers ----
    for (int l = 0; l < NLAYER; ++l) {
        const __bf16* wq  = wqkv_bf + (size_t)l * DMODEL * 3 * DMODEL;
        const __bf16* wo  = wo_bf   + (size_t)l * DMODEL * DMODEL;
        const __bf16* wf1 = w1_bf   + (size_t)l * DMODEL * DFFN;
        const __bf16* wf2 = w2_bf   + (size_t)l * DFFN * DMODEL;

        layernorm_kernel<<<ROWS, 256, 0, stream>>>(
            x, ln1_g + (size_t)l * DMODEL, ln1_b + (size_t)l * DMODEL, xn);

        gemm_bf16_kernel<true, false, false, true>
            <<<dim3(3 * DMODEL / BN, ceildiv(ROWS, BM)), 256, 0, stream>>>(
                xn, wq, b_qkv + (size_t)l * 3 * DMODEL, nullptr,
                nullptr, qkvb, ROWS, 3 * DMODEL, DMODEL);

        attention_kernel<<<BATCH * NHEAD, 256, 0, stream>>>(
            qkvb, attnb, probs, (l == NLAYER - 1) ? 1 : 0);

        gemm_bf16_kernel<true, false, true, false>
            <<<dim3(DMODEL / BN, ceildiv(ROWS, BM)), 256, 0, stream>>>(
                attnb, wo, b_o + (size_t)l * DMODEL, x,
                x, nullptr, ROWS, DMODEL, DMODEL);

        layernorm_kernel<<<ROWS, 256, 0, stream>>>(
            x, ln2_g + (size_t)l * DMODEL, ln2_b + (size_t)l * DMODEL, xn);

        gemm_bf16_kernel<true, true, false, true>
            <<<dim3(DFFN / BN, ceildiv(ROWS, BM)), 256, 0, stream>>>(
                xn, wf1, b1 + (size_t)l * DFFN, nullptr,
                nullptr, hb, ROWS, DFFN, DMODEL);

        gemm_bf16_kernel<true, false, true, false>
            <<<dim3(DMODEL / BN, ceildiv(ROWS, BM)), 256, 0, stream>>>(
                hb, wf2, b2 + (size_t)l * DMODEL, x,
                x, nullptr, ROWS, DMODEL, DFFN);
    }

    // ---- ln_post, cls projection, normalize, cls-attention mean ----
    layernorm_kernel<<<ROWS, 256, 0, stream>>>(x, ln_post_g, ln_post_b, xn);
    gather_cls_kernel<<<BATCH, 256, 0, stream>>>(xn, acls);
    gemm_bf16_kernel<false, false, false, false>
        <<<dim3(OUTD / BN, 1), 256, 0, stream>>>(
            acls, wproj_bf, nullptr, nullptr, feats, nullptr,
            BATCH, OUTD, DMODEL);
    normalize_kernel<<<BATCH, 256, 0, stream>>>(feats);
    head_mean_kernel<<<ceildiv(BATCH * NPATCH, 256), 256, 0, stream>>>(probs, cls_attn);
}
